// DensityMatrixMLP_63428077027605
// MI455X (gfx1250) — matrix operand order repaired and checked by reading, not compiled, not run on hardware
//
#include <hip/hip_runtime.h>
#include <hip/hip_bf16.h>

typedef __attribute__((ext_vector_type(16))) _Float16 v16h;
typedef __attribute__((ext_vector_type(8)))  _Float16 v8h;
typedef __attribute__((ext_vector_type(8)))  float    v8f;
typedef __attribute__((ext_vector_type(4)))  float    v4f;
typedef __attribute__((ext_vector_type(2)))  float    v2f;

#define IN_DIM   256
#define HIDDEN   128
#define TRILN    136
#define NT2      9             // ceil(136/16) output tiles of GEMM2
#define W1T_STR  264           // halfs/row: 256 + 8 pad -> lane stride 132 dwords (4 banks)
#define W2T_STR  136           // halfs/row: 128 + 8 pad -> lane stride 68 dwords (4 banks)
#define H_STR    136           // halfs/row: 128 + 8 pad
#define L_STR    17            // floats/row: 16 + 1 pad
#define L_BUF_F  304           // 16*17 + 32 dump slots (floats) per L buffer

#define LDS_W1T  0
#define LDS_W2T  (128 * W1T_STR * 2)                 // 67584
#define LDS_H    (LDS_W2T + 144 * W2T_STR * 2)       // 106752
#define LDS_L    (LDS_H + 8 * 16 * H_STR * 2)        // 141568
#define LDS_TOT  (LDS_L + 8 * 2 * L_BUF_F * 4)       // 161024 bytes (2 blocks / WGP)

__global__ __launch_bounds__(256)
void density_mlp_kernel(const float* __restrict__ x,  const float* __restrict__ W1,
                        const float* __restrict__ b1, const float* __restrict__ W2,
                        const float* __restrict__ b2, float* __restrict__ out)
{
  extern __shared__ char smem[];
  _Float16* w1t = (_Float16*)(smem + LDS_W1T);   // [128][W1T_STR]  W1^T in f16
  _Float16* w2t = (_Float16*)(smem + LDS_W2T);   // [144][W2T_STR]  W2^T in f16 (cols>=136 zero)

  const int tid  = threadIdx.x;
  const int lane = tid & 31;
  const int w    = tid >> 5;
  const int half = lane >> 4;          // 0: lanes 0-15, 1: lanes 16-31
  const int ln   = lane & 15;
  const int sA   = half * 8;           // 16-bit A-frag K offset for this lane-half
  const int sB   = half * 16;          // unused: B frags take sA, same K order as A

  _Float16* hbuf = (_Float16*)(smem + LDS_H) + w * (16 * H_STR);  // per-wave h tile
  float*    Lbuf = (float*)(smem + LDS_L) + w * (2 * L_BUF_F);    // per-wave 2x dense L

  // ------- Stage W1^T: linear (perfectly coalesced) float4 reads, b16 transpose scatter ------
  #pragma unroll 2
  for (int it = 0; it < 32; ++it) {
    int q4 = (it * 256 + tid) * 4;               // 32768 elements total
    v4f v = *(const v4f*)(W1 + q4);
    #pragma unroll
    for (int j = 0; j < 4; ++j) {
      int q = q4 + j;
      int k = q >> 7, n = q & 127;               // W1[k][n], row-major
      w1t[n * W1T_STR + k] = (_Float16)v[j];
    }
  }
  // ------- Stage W2^T: linear float4 reads (17408 = 17*1024 elements), b16 scatter -----------
  #pragma unroll 2
  for (int it = 0; it < 17; ++it) {
    int q4 = (it * 256 + tid) * 4;
    v4f v = *(const v4f*)(W2 + q4);
    #pragma unroll
    for (int j = 0; j < 4; ++j) {
      int q = q4 + j;
      int k = q / TRILN, n = q - k * TRILN;      // W2[k][n]
      w2t[n * W2T_STR + k] = (_Float16)v[j];
    }
  }
  // zero the pad columns n = 136..143
  for (int q = tid; q < 8 * HIDDEN; q += 256) {
    int n = TRILN + (q >> 7), k = q & 127;
    w2t[n * W2T_STR + k] = (_Float16)0.f;
  }
  __syncthreads();

  const int m0 = (blockIdx.x * 8 + w) * 16;      // this wave's 16 batch rows

  auto loadB1 = [&](int c, int t) -> v16h {
    const _Float16* bp = &w1t[(16 * t + ln) * W1T_STR + 32 * c + sA];
    v8h blo = *(const v8h*)bp;
    v8h bhi = *(const v8h*)(bp + 16);
    v16h bf;
    #pragma unroll
    for (int e = 0; e < 8; ++e) { bf[e] = blo[e]; bf[8 + e] = bhi[e]; }
    return bf;
  };
  auto loadB2 = [&](int c, int t) -> v16h {
    const _Float16* bp = &w2t[(16 * t + ln) * W2T_STR + 32 * c + sA];
    v8h blo = *(const v8h*)bp;
    v8h bhi = *(const v8h*)(bp + 16);
    v16h bf;
    #pragma unroll
    for (int e = 0; e < 8; ++e) { bf[e] = blo[e]; bf[8 + e] = bhi[e]; }
    return bf;
  };

  // ---------------- GEMM1: h[16,128] = x[16,256] @ W1 (f16 WMMA, f32 accum) -------------------
  // Pipeline: 16 groups of 4 fragments (G = 2c + g), ping-pong buffers so group G+1's
  // 8 ds_load_b128 are in flight while group G's 4 WMMAs execute.
  v8f acc1[8];
  #pragma unroll
  for (int t = 0; t < 8; ++t)
    #pragma unroll
    for (int e = 0; e < 8; ++e) acc1[t][e] = 0.f;

  const float* xrow = x + (size_t)(m0 + ln) * IN_DIM;

  v16h bufA[4], bufB[4];
  #pragma unroll
  for (int u = 0; u < 4; ++u) bufA[u] = loadB1(0, u);   // prologue: group 0

  v16h a;
  #pragma unroll
  for (int G = 0; G < 16; ++G) {
    const int c  = G >> 1;
    const int tb = (G & 1) * 4;
    if ((G & 1) == 0) {
      // build A fragment for chunk c: lane row (m0+ln), K in {32c+sA+0..7}++{32c+sA+16..23}
      v4f f0 = *(const v4f*)(xrow + 32 * c + sA);
      v4f f1 = *(const v4f*)(xrow + 32 * c + sA + 4);
      v4f f2 = *(const v4f*)(xrow + 32 * c + sA + 16);
      v4f f3 = *(const v4f*)(xrow + 32 * c + sA + 20);
      #pragma unroll
      for (int e = 0; e < 4; ++e) {
        a[e]      = (_Float16)f0[e];
        a[4 + e]  = (_Float16)f1[e];
        a[8 + e]  = (_Float16)f2[e];
        a[12 + e] = (_Float16)f3[e];
      }
    }
    v16h* curb = (G & 1) ? bufB : bufA;
    v16h* nxtb = (G & 1) ? bufA : bufB;
    if (G < 15) {
      const int cN = (G + 1) >> 1, tbN = ((G + 1) & 1) * 4;
      #pragma unroll
      for (int u = 0; u < 4; ++u) nxtb[u] = loadB1(cN, tbN + u);
    }
    #pragma unroll
    for (int u = 0; u < 4; ++u)
      acc1[tb + u] = __builtin_amdgcn_wmma_f32_16x16x32_f16(false, a, false, curb[u],
                                                            (short)0, acc1[tb + u],
                                                            false, false);
  }

  // ---------------- bias + ReLU, write h tile to per-wave LDS (f16) ---------------------------
  #pragma unroll
  for (int t = 0; t < 8; ++t) {
    float bb = b1[16 * t + ln];
    #pragma unroll
    for (int r = 0; r < 8; ++r) {
      float hv = acc1[t][r] + bb;
      hv = hv > 0.f ? hv : 0.f;
      hbuf[(r + half * 8) * H_STR + 16 * t + ln] = (_Float16)hv;
    }
  }

  // ---------------- GEMM2: v[16,136] = h[16,128] @ W2 ----------------------------------------
  // 12 groups of 3 fragments (G = 3c + g), same ping-pong scheme.
  v8f acc2[NT2];
  #pragma unroll
  for (int t = 0; t < NT2; ++t)
    #pragma unroll
    for (int e = 0; e < 8; ++e) acc2[t][e] = 0.f;

  v16h buf2A[3], buf2B[3];
  #pragma unroll
  for (int u = 0; u < 3; ++u) buf2A[u] = loadB2(0, u);

  #pragma unroll
  for (int G = 0; G < 12; ++G) {
    const int c  = G / 3;
    const int tb = (G % 3) * 3;
    if (tb == 0) {
      const _Float16* ap = &hbuf[ln * H_STR + 32 * c + sA];
      v8h alo = *(const v8h*)ap;                 // K = 32c+sA    .. +7
      v8h ahi = *(const v8h*)(ap + 16);          // K = 32c+sA+16 .. +23
      #pragma unroll
      for (int e = 0; e < 8; ++e) { a[e] = alo[e]; a[8 + e] = ahi[e]; }
    }
    v16h* curb = (G & 1) ? buf2B : buf2A;
    v16h* nxtb = (G & 1) ? buf2A : buf2B;
    if (G < 11) {
      const int cN = (G + 1) / 3, tbN = ((G + 1) % 3) * 3;
      #pragma unroll
      for (int u = 0; u < 3; ++u) nxtb[u] = loadB2(cN, tbN + u);
    }
    #pragma unroll
    for (int u = 0; u < 3; ++u)
      acc2[tb + u] = __builtin_amdgcn_wmma_f32_16x16x32_f16(false, a, false, curb[u],
                                                            (short)0, acc2[tb + u],
                                                            false, false);
  }

  // ---------------- bias b2 -------------------------------------------------------------------
  #pragma unroll
  for (int t = 0; t < NT2; ++t) {
    int n = 16 * t + ln;
    float bb = (n < TRILN) ? b2[n] : 0.f;
    #pragma unroll
    for (int r = 0; r < 8; ++r) acc2[t][r] += bb;
  }

  // ------- traces: trace(L L^T) = ||v||^2, computed once via 16-lane butterfly ---------------
  v8f ssq;
  #pragma unroll
  for (int e = 0; e < 8; ++e) ssq[e] = 0.f;
  #pragma unroll
  for (int t = 0; t < NT2; ++t) {
    bool valid = (16 * t + ln) < TRILN;
    #pragma unroll
    for (int r = 0; r < 8; ++r) {
      float v = acc2[t][r];
      ssq[r] += valid ? v * v : 0.f;
    }
  }
  #pragma unroll
  for (int off = 8; off >= 1; off >>= 1)
    #pragma unroll
    for (int r = 0; r < 8; ++r)
      ssq[r] += __shfl_xor(ssq[r], off, 32);
  // lanes 0-15 hold traces of rows 0..7 in ssq[0..7]; lanes 16-31 hold rows 8..15

  // ------- tril index j -> (i,k): precompute per lane (j = 16t+ln is lane-constant) ----------
  int off_map[NT2];
  #pragma unroll
  for (int t = 0; t < NT2; ++t) {
    int j = 16 * t + ln;
    int i = (int)((__builtin_sqrtf(8.f * (float)j + 1.f) - 1.f) * 0.5f);
    while ((i + 1) * (i + 2) / 2 <= j) ++i;     // fix float rounding
    while (i * (i + 1) / 2 > j) --i;
    off_map[t] = i * L_STR + (j - i * (i + 1) / 2);   // dense-L offset for valid lanes
  }

  // zero both L buffers once (upper triangle stays zero for all 16 rows)
  for (int q = lane; q < 2 * L_BUF_F; q += 32) Lbuf[q] = 0.f;

  const int dump = 16 * L_STR + lane;            // per-lane dump slot inside each buffer

  // unconditional scatter of row m's tril values into dense L (invalid lanes -> dump slot)
  #define SCATTER_ROW(mm, Lb)                                                     \
    {                                                                             \
      const bool mine_ = (half == ((mm) >> 3));                                   \
      _Pragma("unroll")                                                           \
      for (int t = 0; t < NT2; ++t) {                                             \
        bool ok_ = mine_ && ((16 * t + ln) < TRILN);                              \
        (Lb)[ok_ ? off_map[t] : dump] = acc2[t][(mm) & 7];                        \
      }                                                                           \
    }

  SCATTER_ROW(0, Lbuf)                           // prologue into buffer 0

  // ---------------- per batch row: rho = L @ L^T via f32 WMMA, scale, store -------------------
  #pragma unroll
  for (int m = 0; m < 16; ++m) {
    float* cur = Lbuf + (m & 1) * L_BUF_F;

    // issue all fragment loads for row m first (A-frag == B-frag for a Gram product)
    v2f fr[4];
    #pragma unroll
    for (int c = 0; c < 4; ++c) {
      int col = 4 * c + half * 2;                // f32 16x4 A: lane row = ln, K = col, col+1
      fr[c][0] = cur[ln * L_STR + col];
      fr[c][1] = cur[ln * L_STR + col + 1];
    }
    // overlap: scatter row m+1 into the other buffer while row m's WMMAs run
    if (m < 15) SCATTER_ROW(m + 1, Lbuf + ((m + 1) & 1) * L_BUF_F)

    v8f d;
    #pragma unroll
    for (int e = 0; e < 8; ++e) d[e] = 0.f;
    #pragma unroll
    for (int c = 0; c < 4; ++c)
      d = __builtin_amdgcn_wmma_f32_16x16x4_f32(false, fr[c], false, fr[c],
                                                (short)0, d, false, false);

    float tr = __builtin_bit_cast(float,
                 __builtin_amdgcn_readlane(__builtin_bit_cast(int, ssq[m & 7]),
                                           (m >> 3) << 4));
    float invTr = 1.f / tr;

    float* orow = out + (size_t)(m0 + m) * 256;
    #pragma unroll
    for (int r2 = 0; r2 < 8; ++r2)
      orow[(r2 + half * 8) * 16 + ln] = d[r2] * invTr;
  }
  #undef SCATTER_ROW
}

extern "C" void kernel_launch(void* const* d_in, const int* in_sizes, int n_in,
                              void* d_out, int out_size, void* d_ws, size_t ws_size,
                              hipStream_t stream) {
  const float* x  = (const float*)d_in[0];
  const float* W1 = (const float*)d_in[1];
  const float* b1 = (const float*)d_in[2];
  const float* W2 = (const float*)d_in[3];
  const float* b2 = (const float*)d_in[4];
  float* out = (float*)d_out;

  const int batch  = in_sizes[0] / IN_DIM;     // 131072
  const int blocks = batch / 128;              // 128 rows per block (8 waves x 16)

  hipLaunchKernelGGL(density_mlp_kernel, dim3(blocks), dim3(256), LDS_TOT, stream,
                     x, W1, b1, W2, b2, out);
}
